// QuantumAttention_20306605375887
// MI455X (gfx1250) — matrix operand order repaired and checked by reading, not compiled, not run on hardware
//
#include <hip/hip_runtime.h>
#include <hip/hip_bf16.h>

// ---------------------------------------------------------------------------
// QuantumAttention on gfx1250 (MI455X, wave32, WMMA bf16 + async-to-LDS DMA).
//
// scores = Qc Kc^T / 8 + Q E K^T  ==  Q (diag(cos^2 phi / 8) + E) K^T
// Fold M_h = diag(cos^2/8)+E into the Q projection weights, then it's plain
// flash attention + 4 GEMMs, all on v_wmma_f32_16x16x32_bf16, with
// double-buffered LDS tiles filled by global_load_async_to_lds_b128.
// ---------------------------------------------------------------------------

typedef __attribute__((ext_vector_type(8)))  __bf16 bf16x8;
typedef __attribute__((ext_vector_type(16))) __bf16 bf16x16;
typedef __attribute__((ext_vector_type(8)))  float  f32x8;
typedef __attribute__((ext_vector_type(4)))  int    v4i;

#define EMBED 1024
#define SEQ   2048
#define BATCH 2
#define NH    16
#define HD    64
#define ROWS  (BATCH*SEQ)   // 4096

#if defined(__AMDGCN__) && __has_builtin(__builtin_amdgcn_global_load_async_to_lds_b128)
#define HAVE_ASYNC_LDS 1
#endif

#if defined(__AMDGCN__)
typedef __attribute__((address_space(1))) v4i* gptr_v4i;   // global int4*
typedef __attribute__((address_space(3))) v4i* lptr_v4i;   // LDS int4*
#endif

// 16-byte global -> LDS copy. Async DMA (ASYNCcnt) when available, else a
// plain generic copy (global_load + ds_store).
__device__ __forceinline__ void copy_b128(const __bf16* g, __bf16* l) {
#if defined(HAVE_ASYNC_LDS)
  __builtin_amdgcn_global_load_async_to_lds_b128((gptr_v4i)g, (lptr_v4i)l, 0, 0);
#else
  *(float4*)l = *(const float4*)g;
#endif
}

__device__ __forceinline__ void wait_async() {
#if defined(HAVE_ASYNC_LDS)
#if __has_builtin(__builtin_amdgcn_s_wait_asynccnt)
  __builtin_amdgcn_s_wait_asynccnt(0);
#else
  asm volatile("s_wait_asynccnt 0" ::: "memory");
#endif
#endif
}

// ---- WMMA helpers ---------------------------------------------------------

__device__ __forceinline__ f32x8 wmma_bf16(bf16x16 a, bf16x16 b, f32x8 c) {
  return __builtin_amdgcn_wmma_f32_16x16x32_bf16(false, a, false, b,
                                                 (short)0, c, false, false);
}

// A-operand fragment (16x32, M x K).  ISA 7.12.2: lanes<16 hold K
// {0..7,16..23}, lanes>=16 hold K {8..15,24..31}.
__device__ __forceinline__ bf16x16 load_frag_a(const __bf16* base, int row,
                                               int k0, int ldk) {
  const int lane  = threadIdx.x & 31;
  const int khalf = (lane & 16) ? 8 : 0;
  const __bf16* p = base + (size_t)row * ldk + k0 + khalf;
  bf16x8 lo = *(const bf16x8*)p;          // K = khalf .. khalf+7
  bf16x8 hi = *(const bf16x8*)(p + 16);   // K = khalf+16 .. +23
  bf16x16 f;
#pragma unroll
  for (int i = 0; i < 8; ++i) { f[i] = lo[i]; f[i + 8] = hi[i]; }
  return f;
}

// B-operand fragment (32x16, K x N).  lane = column n; lanes<16 hold K
// {0..7,16..23}, lanes>=16 hold K {8..15,24..31}.  B stored N-major.
__device__ __forceinline__ bf16x16 load_frag_b(const __bf16* base, int col,
                                               int k0, int ldk) {
  const int lane = threadIdx.x & 31;
  const int koff = (lane & 16) ? 8 : 0;
  const __bf16* p = base + (size_t)col * ldk + k0 + koff;
  bf16x8 lo = *(const bf16x8*)p;
  bf16x8 hi = *(const bf16x8*)(p + 16);
  bf16x16 f;
#pragma unroll
  for (int i = 0; i < 8; ++i) { f[i] = lo[i]; f[i + 8] = hi[i]; }
  return f;
}

// ---- prep kernels ---------------------------------------------------------

__global__ void cvt_f32_to_bf16(const float* __restrict__ in,
                                __bf16* __restrict__ out, int n) {
  int i = blockIdx.x * blockDim.x + threadIdx.x;
  int stride = gridDim.x * blockDim.x;
  for (; i < n; i += stride) out[i] = (__bf16)in[i];
}

// w: [K=1024][N=1024] fp32  ->  wt: [N][K] bf16 (so B-frag reads contiguous k)
__global__ void cvt_transpose_w(const float* __restrict__ w,
                                __bf16* __restrict__ wt) {
  int i = blockIdx.x * blockDim.x + threadIdx.x;
  int stride = gridDim.x * blockDim.x;
  for (; i < EMBED * EMBED; i += stride) {
    int n = i >> 10, k = i & 1023;
    wt[i] = (__bf16)w[k * EMBED + n];
  }
}

// Fold M_h = diag(cos^2(phase)/8) + E into q_w / q_b.
// Output Wqt[n][k] (transposed bf16) and bq[n] (f32).  n = h*64 + e.
__global__ void build_wq(const float* __restrict__ qw,
                         const float* __restrict__ qb,
                         const float* __restrict__ phase,
                         const float* __restrict__ ent,
                         __bf16* __restrict__ Wqt, float* __restrict__ bq) {
  const int n = blockIdx.x;
  const int h = n >> 6, e = n & 63;
  __shared__ float mcol[HD];
  if (threadIdx.x < HD) {
    int d = threadIdx.x;
    float v = ent[((size_t)h * HD + d) * HD + e];
    if (d == e) {
      float c = cosf(phase[h * HD + d]);
      v += c * c * 0.125f;
    }
    mcol[d] = v;
  }
  __syncthreads();
  for (int k = threadIdx.x; k < EMBED; k += blockDim.x) {
    float s = 0.f;
#pragma unroll
    for (int d = 0; d < HD; ++d) s += qw[(size_t)k * EMBED + h * HD + d] * mcol[d];
    Wqt[(size_t)n * EMBED + k] = (__bf16)s;
  }
  if (threadIdx.x == 0) {
    float s = 0.f;
    for (int d = 0; d < HD; ++d) s += qb[h * HD + d] * mcol[d];
    bq[n] = s;
  }
}

// ---- tiled WMMA GEMM:  C[4096,1024] = A[4096,1024] x W[1024,1024] + bias --
// 128x128 tile, 8 waves; wave w owns rows w*16..+15 and all 128 cols.
// Double-buffered LDS tiles, async-to-LDS prefetch of the next k-slab.
// mode 0/1: write Q'/K -> [b,h,s,d] bf16
// mode 2:   write V    -> [b,h,d,s] bf16 (transposed for the PV B-operand)
// mode 3:   write f32 row-major (final output)
__global__ __launch_bounds__(256) void gemm_proj(const __bf16* __restrict__ A,
                                                 const __bf16* __restrict__ Bt,
                                                 const float* __restrict__ bias,
                                                 void* __restrict__ out,
                                                 int mode) {
  __shared__ __bf16 As[2][128 * 40];   // ld=40 elems (80B, 16B aligned)
  __shared__ __bf16 Bs[2][128 * 40];
  const int lane = threadIdx.x & 31;
  const int wave = threadIdx.x >> 5;
  const int l15  = lane & 15;
  const int roff = (lane >> 4) * 8;
  const int m0 = blockIdx.y * 128;
  const int n0 = blockIdx.x * 128;
  const int row  = threadIdx.x >> 1;   // staging: 0..127
  const int half = threadIdx.x & 1;    // 16 bf16 (32B) per thread per tile

  f32x8 acc[8];
#pragma unroll
  for (int t = 0; t < 8; ++t)
#pragma unroll
    for (int i = 0; i < 8; ++i) acc[t][i] = 0.f;

  // stage k-slab k0 into buffer b
  auto stage = [&](int k0, int b) {
    const __bf16* ga = A + (size_t)(m0 + row) * EMBED + k0 + half * 16;
    __bf16* la = &As[b][row * 40 + half * 16];
    copy_b128(ga, la);
    copy_b128(ga + 8, la + 8);
    const __bf16* gb = Bt + (size_t)(n0 + row) * EMBED + k0 + half * 16;
    __bf16* lb = &Bs[b][row * 40 + half * 16];
    copy_b128(gb, lb);
    copy_b128(gb + 8, lb + 8);
  };

  int p = 0;
  stage(0, 0);
  for (int k0 = 0; k0 < EMBED; k0 += 32, p ^= 1) {
    wait_async();
    __syncthreads();                 // tile p landed; prior reads of p^1 done
    if (k0 + 32 < EMBED) stage(k0 + 32, p ^ 1);   // DMA overlaps WMMAs below
    bf16x16 a = load_frag_a(&As[p][0], wave * 16 + l15, 0, 40);
#pragma unroll
    for (int nt = 0; nt < 8; ++nt) {
      bf16x16 b = load_frag_b(&Bs[p][0], nt * 16 + l15, 0, 40);
      acc[nt] = wmma_bf16(a, b, acc[nt]);
    }
  }

  // epilogue: C layout -> row = vgpr + roff, col = lane&15
#pragma unroll
  for (int nt = 0; nt < 8; ++nt) {
    int gn = n0 + nt * 16 + l15;
    float bv = bias[gn];
#pragma unroll
    for (int i = 0; i < 8; ++i) {
      int gm = m0 + wave * 16 + roff + i;
      float v = acc[nt][i] + bv;
      if (mode == 3) {
        ((float*)out)[(size_t)gm * EMBED + gn] = v;
      } else {
        int b = gm >> 11, s = gm & (SEQ - 1);
        int h = gn >> 6,  d = gn & 63;
        __bf16 bf = (__bf16)v;
        if (mode == 2)
          ((__bf16*)out)[((size_t)(b * NH + h) * HD + d) * SEQ + s] = bf;
        else
          ((__bf16*)out)[((size_t)(b * NH + h) * SEQ + s) * HD + d] = bf;
      }
    }
  }
}

// ---- flash attention ------------------------------------------------------
// grid (qchunk=16, bh=32), 256 threads (8 waves), wave owns 16 query rows.
// Q'/K: [bh][s][64] bf16, Vt: [bh][64][s] bf16, out -> attended [b,s,h*64+d].
// K/V tiles double-buffered in LDS, filled by async-to-LDS DMA.
__global__ __launch_bounds__(256) void flash_attn(const __bf16* __restrict__ Q,
                                                  const __bf16* __restrict__ Kg,
                                                  const __bf16* __restrict__ Vt,
                                                  __bf16* __restrict__ O) {
  __shared__ __bf16 Ks[2][64 * 72];    // [key][dim], ld 72 (144B, 16B-aligned)
  __shared__ __bf16 Vs[2][64 * 72];    // [dim][key]
  __shared__ __bf16 Ps[8 * 16 * 64];   // per-wave P staging, ld 64
  const int lane = threadIdx.x & 31;
  const int wave = threadIdx.x >> 5;
  const int l15  = lane & 15;
  const int roff = (lane >> 4) * 8;
  const int bh = blockIdx.y;
  const int q0 = blockIdx.x * 128 + wave * 16;
  const size_t qkbase = (size_t)bh * SEQ * HD;
  const size_t vbase  = (size_t)bh * HD * SEQ;
  const int srow  = threadIdx.x >> 2;  // staging: 0..63
  const int spart = threadIdx.x & 3;   // 16 bf16 per thread per tile

  const __bf16* qrow = Q + qkbase + (size_t)q0 * HD;
  bf16x16 qf0 = load_frag_a(qrow, l15, 0, HD);
  bf16x16 qf1 = load_frag_a(qrow, l15, 32, HD);

  f32x8 acc[4];
  float m[8], l[8];
#pragma unroll
  for (int t = 0; t < 4; ++t)
#pragma unroll
    for (int i = 0; i < 8; ++i) acc[t][i] = 0.f;
#pragma unroll
  for (int i = 0; i < 8; ++i) { m[i] = -1e30f; l[i] = 0.f; }

  auto stage_kv = [&](int j, int b) {
    const __bf16* gk = Kg + qkbase + (size_t)(j + srow) * HD + spart * 16;
    __bf16* lk = &Ks[b][srow * 72 + spart * 16];
    copy_b128(gk, lk);
    copy_b128(gk + 8, lk + 8);
    const __bf16* gv = Vt + vbase + (size_t)srow * SEQ + j + spart * 16;
    __bf16* lv = &Vs[b][srow * 72 + spart * 16];
    copy_b128(gv, lv);
    copy_b128(gv + 8, lv + 8);
  };

  int p = 0;
  stage_kv(0, 0);
  for (int j = 0; j < SEQ; j += 64, p ^= 1) {
    wait_async();
    __syncthreads();                 // K/V tile p landed; reads of p^1 done
    if (j + 64 < SEQ) stage_kv(j + 64, p ^ 1);   // DMA overlaps compute

    // S = Q' K^T  (16 x 64 block, f32)
    f32x8 sv[4];
#pragma unroll
    for (int nt = 0; nt < 4; ++nt) {
      f32x8 c = {0.f, 0.f, 0.f, 0.f, 0.f, 0.f, 0.f, 0.f};
      bf16x16 b0 = load_frag_b(&Ks[p][0], nt * 16 + l15, 0, 72);
      c = wmma_bf16(qf0, b0, c);
      bf16x16 b1 = load_frag_b(&Ks[p][0], nt * 16 + l15, 32, 72);
      c = wmma_bf16(qf1, b1, c);
      sv[nt] = c;
    }

    // online softmax: row r lives across cols 0..15 of one 16-lane half
    float mnew[8], scale[8];
#pragma unroll
    for (int i = 0; i < 8; ++i) {
      float mm = fmaxf(fmaxf(sv[0][i], sv[1][i]), fmaxf(sv[2][i], sv[3][i]));
      mm = fmaxf(mm, __shfl_xor(mm, 1, 32));
      mm = fmaxf(mm, __shfl_xor(mm, 2, 32));
      mm = fmaxf(mm, __shfl_xor(mm, 4, 32));
      mm = fmaxf(mm, __shfl_xor(mm, 8, 32));
      mnew[i]  = fmaxf(m[i], mm);
      scale[i] = __expf(m[i] - mnew[i]);
      m[i] = mnew[i];
    }
#pragma unroll
    for (int nt = 0; nt < 4; ++nt)
#pragma unroll
      for (int i = 0; i < 8; ++i) sv[nt][i] = __expf(sv[nt][i] - mnew[i]);
#pragma unroll
    for (int i = 0; i < 8; ++i) {
      float ls = sv[0][i] + sv[1][i] + sv[2][i] + sv[3][i];
      ls += __shfl_xor(ls, 1, 32);
      ls += __shfl_xor(ls, 2, 32);
      ls += __shfl_xor(ls, 4, 32);
      ls += __shfl_xor(ls, 8, 32);
      l[i] = l[i] * scale[i] + ls;
    }
#pragma unroll
    for (int dt = 0; dt < 4; ++dt)
#pragma unroll
      for (int i = 0; i < 8; ++i) acc[dt][i] *= scale[i];

    // stage P (C layout) -> LDS -> reload as A fragments (bf16)
    __bf16* pw = Ps + wave * 16 * 64;
#pragma unroll
    for (int nt = 0; nt < 4; ++nt)
#pragma unroll
      for (int i = 0; i < 8; ++i)
        pw[(roff + i) * 64 + nt * 16 + l15] = (__bf16)sv[nt][i];
    asm volatile("s_wait_dscnt 0" ::: "memory");   // CDNA5 split DS counter
    bf16x16 pf0 = load_frag_a(pw, l15, 0, 64);
    bf16x16 pf1 = load_frag_a(pw, l15, 32, 64);

    // acc += P V   (B[k=key][n=dim] = Vt[dim][key], contiguous per lane)
#pragma unroll
    for (int dt = 0; dt < 4; ++dt) {
      bf16x16 v0 = load_frag_b(&Vs[p][0], dt * 16 + l15, 0, 72);
      acc[dt] = wmma_bf16(pf0, v0, acc[dt]);
      bf16x16 v1 = load_frag_b(&Vs[p][0], dt * 16 + l15, 32, 72);
      acc[dt] = wmma_bf16(pf1, v1, acc[dt]);
    }
  }

  // normalize and write attended as [b, s, h*64+d] bf16 (row-major 4096x1024)
  const int b = bh >> 4, h = bh & 15;
#pragma unroll
  for (int i = 0; i < 8; ++i) {
    float inv = 1.0f / l[i];
    int sq = q0 + roff + i;
    size_t rb = ((size_t)(b * SEQ + sq)) * EMBED + h * HD;
#pragma unroll
    for (int dt = 0; dt < 4; ++dt)
      O[rb + dt * 16 + l15] = (__bf16)(acc[dt][i] * inv);
  }
}

// ---- launch ---------------------------------------------------------------

extern "C" void kernel_launch(void* const* d_in, const int* in_sizes, int n_in,
                              void* d_out, int out_size, void* d_ws, size_t ws_size,
                              hipStream_t stream) {
  (void)in_sizes; (void)n_in; (void)out_size; (void)ws_size;
  const float* x     = (const float*)d_in[0];
  const float* q_w   = (const float*)d_in[1];
  const float* q_b   = (const float*)d_in[2];
  const float* k_w   = (const float*)d_in[3];
  const float* k_b   = (const float*)d_in[4];
  const float* v_w   = (const float*)d_in[5];
  const float* v_b   = (const float*)d_in[6];
  const float* phase = (const float*)d_in[7];
  const float* ent   = (const float*)d_in[8];
  const float* o_w   = (const float*)d_in[9];
  const float* o_b   = (const float*)d_in[10];

  char* ws = (char*)d_ws;
  size_t off = 0;
  __bf16* Xbf = (__bf16*)(ws + off); off += (size_t)ROWS * EMBED * 2;   // 8 MB
  __bf16* Wqt = (__bf16*)(ws + off); off += (size_t)EMBED * EMBED * 2;  // 2 MB
  __bf16* Wkt = (__bf16*)(ws + off); off += (size_t)EMBED * EMBED * 2;
  __bf16* Wvt = (__bf16*)(ws + off); off += (size_t)EMBED * EMBED * 2;
  __bf16* Wot = (__bf16*)(ws + off); off += (size_t)EMBED * EMBED * 2;
  float*  bq  = (float*)(ws + off);  off += (size_t)EMBED * 4;
  __bf16* Qb  = (__bf16*)(ws + off); off += (size_t)ROWS * EMBED * 2;   // [b,h,s,d]
  __bf16* Kb  = (__bf16*)(ws + off); off += (size_t)ROWS * EMBED * 2;   // [b,h,s,d]
  __bf16* Vtb = (__bf16*)(ws + off); off += (size_t)ROWS * EMBED * 2;   // [b,h,d,s]
  __bf16* Ab  = (__bf16*)(ws + off); off += (size_t)ROWS * EMBED * 2;   // [b,s,e]
  // total ~48 MB of workspace

  cvt_f32_to_bf16<<<1024, 256, 0, stream>>>(x, Xbf, ROWS * EMBED);
  cvt_transpose_w<<<1024, 256, 0, stream>>>(k_w, Wkt);
  cvt_transpose_w<<<1024, 256, 0, stream>>>(v_w, Wvt);
  cvt_transpose_w<<<1024, 256, 0, stream>>>(o_w, Wot);
  build_wq<<<1024, 256, 0, stream>>>(q_w, q_b, phase, ent, Wqt, bq);

  dim3 gg(EMBED / 128, ROWS / 128);   // (8, 32)
  gemm_proj<<<gg, 256, 0, stream>>>(Xbf, Wqt, bq,  (void*)Qb,  0);
  gemm_proj<<<gg, 256, 0, stream>>>(Xbf, Wkt, k_b, (void*)Kb,  1);
  gemm_proj<<<gg, 256, 0, stream>>>(Xbf, Wvt, v_b, (void*)Vtb, 2);

  flash_attn<<<dim3(SEQ / 128, BATCH * NH), 256, 0, stream>>>(Qb, Kb, Vtb, Ab);

  gemm_proj<<<gg, 256, 0, stream>>>(Ab, Wot, o_b, d_out, 3);
}
